// TopKSAE_43482248905073
// MI455X (gfx1250) — matrix operand order repaired and checked by reading, not compiled, not run on hardware
//
#include <hip/hip_runtime.h>
#include <hip/hip_bf16.h>

// Problem constants (match reference)
constexpr int kB = 8192;
constexpr int kD = 1024;
constexpr int kN = 16384;
constexpr int kK = 32;

typedef __bf16 v16bf __attribute__((ext_vector_type(16)));
typedef __bf16 v4bf  __attribute__((ext_vector_type(4)));
typedef float  v8f   __attribute__((ext_vector_type(8)));

// Exact pointee type expected by the gfx1250 async-to-LDS builtins:
// GCC-style vector of 4 ints, in global (as1) / LDS (as3) address spaces.
typedef int v4i_gcc __attribute__((vector_size(4 * sizeof(int))));
typedef __attribute__((address_space(1))) v4i_gcc gv4i;
typedef __attribute__((address_space(3))) v4i_gcc lv4i;

#if defined(__has_builtin)
#  if __has_builtin(__builtin_amdgcn_global_load_async_to_lds_b128)
#    define SAE_ASYNC_LDS 1
#  endif
#endif
#ifndef SAE_ASYNC_LDS
#  define SAE_ASYNC_LDS 0
#endif

#if SAE_ASYNC_LDS
#  if __has_builtin(__builtin_amdgcn_s_wait_asynccnt)
#    define SAE_WAIT_ASYNC() __builtin_amdgcn_s_wait_asynccnt(0)
#  else
#    define SAE_WAIT_ASYNC() asm volatile("s_wait_asynccnt 0x0" ::: "memory")
#  endif
#  define SAE_ASYNC_B128(gp, lp) \
     __builtin_amdgcn_global_load_async_to_lds_b128((gv4i*)(gp), (lv4i*)(lp), 0, 0)
#endif

// ---------------------------------------------------------------------------
// Prep 1: xb = bf16(x - b), vectorized 4 elements/thread
// ---------------------------------------------------------------------------
__global__ __launch_bounds__(256) void sae_prep_xb(
    const float4* __restrict__ x, const float* __restrict__ bv,
    v4bf* __restrict__ xb) {
  size_t i = (size_t)blockIdx.x * 256 + threadIdx.x;   // over B*D/4
  float4 v = x[i];
  size_t d = (i * 4) & (size_t)(kD - 1);               // D power of two
  float4 bb = *(const float4*)(bv + d);
  v4bf o;
  o[0] = (__bf16)(v.x - bb.x);
  o[1] = (__bf16)(v.y - bb.y);
  o[2] = (__bf16)(v.z - bb.z);
  o[3] = (__bf16)(v.w - bb.w);
  xb[i] = o;
}

// ---------------------------------------------------------------------------
// Prep 2: W_enc fp32 -> bf16
// ---------------------------------------------------------------------------
__global__ __launch_bounds__(256) void sae_prep_wenc(
    const float4* __restrict__ we, v4bf* __restrict__ web) {
  size_t i = (size_t)blockIdx.x * 256 + threadIdx.x;   // over N*D/4
  float4 v = we[i];
  v4bf o;
  o[0] = (__bf16)v.x; o[1] = (__bf16)v.y; o[2] = (__bf16)v.z; o[3] = (__bf16)v.w;
  web[i] = o;
}

// ---------------------------------------------------------------------------
// Prep 3: W_decT[n][d] = W_dec[d][n]  (tiled transpose, 32x32 tiles)
// ---------------------------------------------------------------------------
__global__ __launch_bounds__(256) void sae_transpose_wdec(
    const float* __restrict__ wd, float* __restrict__ wdt) {
  __shared__ float tile[32][33];
  const int n0 = blockIdx.x * 32;
  const int d0 = blockIdx.y * 32;
  const int tx = threadIdx.x;     // 0..31
  const int ty = threadIdx.y;     // 0..7
#pragma unroll
  for (int k = 0; k < 32; k += 8)
    tile[ty + k][tx] = wd[(size_t)(d0 + ty + k) * kN + n0 + tx];
  __syncthreads();
#pragma unroll
  for (int k = 0; k < 32; k += 8)
    wdt[(size_t)(n0 + ty + k) * kD + d0 + tx] = tile[tx][ty + k];
}

// ---------------------------------------------------------------------------
// Encoder GEMM: z = relu(xb @ Wenc^T), bf16 inputs, fp32 accumulate via WMMA.
// Block = 8 waves, 128x128 tile; wave = 32x64 (2x4 WMMA tiles); Kc = 32.
// Async-to-LDS double-buffered when the gfx1250 async builtins are available.
// ---------------------------------------------------------------------------
constexpr int LDT = 40;   // LDS row stride in bf16 (64B row + 16B pad, 16B aligned)

__global__ __launch_bounds__(256) void sae_enc_gemm(
    const __bf16* __restrict__ xb, const __bf16* __restrict__ we,
    float* __restrict__ z) {
#if SAE_ASYNC_LDS
  __shared__ __bf16 ldsA[2][128 * LDT];
  __shared__ __bf16 ldsB[2][128 * LDT];
#else
  __shared__ __bf16 ldsA[1][128 * LDT];
  __shared__ __bf16 ldsB[1][128 * LDT];
#endif

  const int tid  = threadIdx.x;
  const int lane = tid & 31;
  const int wid  = tid >> 5;     // 0..7
  const int wm   = wid & 3;      // 4 waves along M (32 rows each)
  const int wn   = wid >> 2;     // 2 waves along N (64 cols each)
  const int lm   = lane & 15;
  const int lh   = lane >> 4;    // lane half, per WMMA VGPR layout
  const int row0 = blockIdx.y * 128;
  const int col0 = blockIdx.x * 128;

  v8f acc[2][4];
#pragma unroll
  for (int i = 0; i < 2; ++i)
#pragma unroll
    for (int j = 0; j < 4; ++j)
      acc[i][j] = (v8f){};

  // cooperative LDS fill: 256 threads x (32B A + 32B B) = 16KB per K-chunk
  const int r = tid >> 1;        // 0..127
  const int h = tid & 1;         // 0..1
  const __bf16* gA = xb + (size_t)(row0 + r) * kD + h * 16;
  const __bf16* gB = we + (size_t)(col0 + r) * kD + h * 16;
  const int ldso = r * LDT + h * 16;

  union Frag { v16bf v; uint4 q[2]; };

#if SAE_ASYNC_LDS
  // -------- async double-buffered pipeline --------
  constexpr int NIT = kD / 32;
  // prologue: fill buffer 0 with K-chunk 0
  SAE_ASYNC_B128(gA,     ldsA[0] + ldso);
  SAE_ASYNC_B128(gA + 8, ldsA[0] + ldso + 8);
  SAE_ASYNC_B128(gB,     ldsB[0] + ldso);
  SAE_ASYNC_B128(gB + 8, ldsB[0] + ldso + 8);

#pragma unroll 1
  for (int it = 0; it < NIT; ++it) {
    const int cur = it & 1;
    SAE_WAIT_ASYNC();           // my async writes into buf[cur] have landed
    __syncthreads();            // => buf[cur] complete; buf[1-cur] reads done
    if (it + 1 < NIT) {
      const int kc = (it + 1) * 32;
      const int nxt = 1 - cur;
      SAE_ASYNC_B128(gA + kc,     ldsA[nxt] + ldso);
      SAE_ASYNC_B128(gA + kc + 8, ldsA[nxt] + ldso + 8);
      SAE_ASYNC_B128(gB + kc,     ldsB[nxt] + ldso);
      SAE_ASYNC_B128(gB + kc + 8, ldsB[nxt] + ldso + 8);
    }

    // fragments from buf[cur]
    Frag af[2], bf_[4];
#pragma unroll
    for (int mt = 0; mt < 2; ++mt) {
      const __bf16* ap = ldsA[cur] + (wm * 32 + mt * 16 + lm) * LDT + lh * 16;
      af[mt].q[0] = *(const uint4*)ap;          // K = lh*16 .. lh*16+7
      af[mt].q[1] = *(const uint4*)(ap + 8);    // K = lh*16+8 ..
    }
#pragma unroll
    for (int nt = 0; nt < 4; ++nt) {
      const __bf16* bp = ldsB[cur] + (wn * 64 + nt * 16 + lm) * LDT + lh * 16;
      bf_[nt].q[0] = *(const uint4*)bp;
      bf_[nt].q[1] = *(const uint4*)(bp + 8);
    }
#pragma unroll
    for (int mt = 0; mt < 2; ++mt)
#pragma unroll
      for (int nt = 0; nt < 4; ++nt)
        acc[mt][nt] = __builtin_amdgcn_wmma_f32_16x16x32_bf16(
            false, af[mt].v, false, bf_[nt].v, (short)0, acc[mt][nt],
            false, false);
  }
#else
  // -------- fallback: register-staged LDS fill --------
  uint4* dA = (uint4*)(ldsA[0] + ldso);
  uint4* dB = (uint4*)(ldsB[0] + ldso);
  for (int kc = 0; kc < kD; kc += 32) {
    const uint4* sA = (const uint4*)(gA + kc);
    const uint4* sB = (const uint4*)(gB + kc);
    uint4 a0 = sA[0], a1 = sA[1];
    uint4 b0 = sB[0], b1 = sB[1];
    if (kc + 32 < kD) {                       // uniform branch
      __builtin_prefetch(gA + kc + 32, 0, 3); // near (WGP-level) prefetch
      __builtin_prefetch(gB + kc + 32, 0, 3);
    }
    __syncthreads();
    dA[0] = a0; dA[1] = a1;
    dB[0] = b0; dB[1] = b1;
    __syncthreads();

    Frag af[2], bf_[4];
#pragma unroll
    for (int mt = 0; mt < 2; ++mt) {
      const __bf16* ap = ldsA[0] + (wm * 32 + mt * 16 + lm) * LDT + lh * 16;
      af[mt].q[0] = *(const uint4*)ap;
      af[mt].q[1] = *(const uint4*)(ap + 8);
    }
#pragma unroll
    for (int nt = 0; nt < 4; ++nt) {
      const __bf16* bp = ldsB[0] + (wn * 64 + nt * 16 + lm) * LDT + lh * 16;
      bf_[nt].q[0] = *(const uint4*)bp;
      bf_[nt].q[1] = *(const uint4*)(bp + 8);
    }
#pragma unroll
    for (int mt = 0; mt < 2; ++mt)
#pragma unroll
      for (int nt = 0; nt < 4; ++nt)
        acc[mt][nt] = __builtin_amdgcn_wmma_f32_16x16x32_bf16(
            false, af[mt].v, false, bf_[nt].v, (short)0, acc[mt][nt],
            false, false);
  }
#endif

  // Store D tiles with fused ReLU.  C/D layout: VGPR r -> M = r + 8*lanehalf, N = lane&15.
#pragma unroll
  for (int mt = 0; mt < 2; ++mt) {
    const int rowb = row0 + wm * 32 + mt * 16 + lh * 8;
#pragma unroll
    for (int nt = 0; nt < 4; ++nt) {
      const int col = col0 + wn * 64 + nt * 16 + lm;
      float* zp = z + (size_t)rowb * kN + col;
#pragma unroll
      for (int rr = 0; rr < 8; ++rr)
        zp[(size_t)rr * kN] = fmaxf(acc[mt][nt][rr], 0.0f);
    }
  }
}

// ---------------------------------------------------------------------------
// Top-K(32) per row via 4-pass radix select on float bits (all values >= 0
// after ReLU, so unsigned bit patterns are monotonic). Deterministic
// compaction via per-thread counts + serial exclusive scan (no atomic-order
// dependence).
// ---------------------------------------------------------------------------
__global__ __launch_bounds__(256) void sae_topk(
    float* __restrict__ z, int* __restrict__ idxw, float* __restrict__ valw,
    int* __restrict__ cntw) {
  const int row = blockIdx.x;
  float* zr = z + (size_t)row * kN;
  __shared__ unsigned hist[256];
  __shared__ unsigned s_sel, s_want;
  __shared__ int cnts[256];
  const int tid = threadIdx.x;

  unsigned prefix = 0;
  unsigned want = kK;
#pragma unroll 1
  for (int shift = 24; shift >= 0; shift -= 8) {
    __syncthreads();
    hist[tid] = 0;
    __syncthreads();
    const unsigned himask = (shift == 24) ? 0u : (~0u << (shift + 8));
    for (int i = tid; i < kN; i += 256) {
      unsigned u = __float_as_uint(zr[i]);
      if ((u & himask) == prefix)
        atomicAdd(&hist[(u >> shift) & 255u], 1u);
    }
    __syncthreads();
    if (tid == 0) {
      unsigned cum = 0, d = 256;
      do { --d; cum += hist[d]; } while (cum < want && d > 0);
      s_sel  = d;
      s_want = want - (cum - hist[d]);
    }
    __syncthreads();
    prefix |= (s_sel << shift);
    want = s_want;
  }
  const unsigned T = prefix;  // bit pattern of the K-th largest value

  // Pass 1: zero everything below threshold; count kept nonzeros per thread.
  int c = 0;
  for (int i = tid; i < kN; i += 256) {
    float v = zr[i];
    unsigned u = __float_as_uint(v);
    if (u < T) zr[i] = 0.0f;
    else if (v > 0.0f) ++c;
  }
  cnts[tid] = c;
  __syncthreads();
  if (tid == 0) {
    int s = 0;
    for (int j = 0; j < 256; ++j) { int t = cnts[j]; cnts[j] = s; s += t; }
    cntw[row] = (s < 2 * kK) ? s : 2 * kK;
  }
  __syncthreads();
  // Pass 2: deterministic compaction of kept (index, value) pairs.
  int off = cnts[tid];
  for (int i = tid; i < kN; i += 256) {
    float v = zr[i];
    unsigned u = __float_as_uint(v);
    if (u >= T && v > 0.0f) {
      if (off < 2 * kK) {
        idxw[(size_t)row * (2 * kK) + off] = i;
        valw[(size_t)row * (2 * kK) + off] = v;
      }
      ++off;
    }
  }
}

// ---------------------------------------------------------------------------
// Sparse decode: recon[row] = b + sum_j val_j * W_decT[idx_j, :]
// W_decT rows are contiguous (4KB) -> coalesced; W_decT (64MB) is L2-resident.
// ---------------------------------------------------------------------------
__global__ __launch_bounds__(256) void sae_recon(
    const float* __restrict__ wdt, const float* __restrict__ bv,
    const int* __restrict__ idxw, const float* __restrict__ valw,
    const int* __restrict__ cntw, float* __restrict__ recon) {
  const int row = blockIdx.x;
  const int tid = threadIdx.x;
  __shared__ int   s_idx[2 * kK];
  __shared__ float s_val[2 * kK];
  const int m = cntw[row];
  if (tid < 2 * kK && tid < m) {
    s_idx[tid] = idxw[(size_t)row * (2 * kK) + tid];
    s_val[tid] = valw[(size_t)row * (2 * kK) + tid];
  }
  __syncthreads();
#pragma unroll 1
  for (int d = tid; d < kD; d += 256) {
    float acc = bv[d];
    for (int j = 0; j < m; ++j)
      acc += s_val[j] * wdt[(size_t)s_idx[j] * kD + d];
    recon[(size_t)row * kD + d] = acc;
  }
}

// ---------------------------------------------------------------------------
// Launch
// ---------------------------------------------------------------------------
extern "C" void kernel_launch(void* const* d_in, const int* in_sizes, int n_in,
                              void* d_out, int out_size, void* d_ws, size_t ws_size,
                              hipStream_t stream) {
  const float* x  = (const float*)d_in[0];   // (B, D)
  const float* We = (const float*)d_in[1];   // (N, D)
  const float* Wd = (const float*)d_in[2];   // (D, N)
  const float* bv = (const float*)d_in[3];   // (D,)

  float* recon = (float*)d_out;                         // (B, D)
  float* z     = (float*)d_out + (size_t)kB * kD;       // (B, N)

  // Workspace layout
  char* ws = (char*)d_ws;
  const size_t XB_BYTES  = (size_t)kB * kD * 2;   // 16 MB  bf16(x - b)
  const size_t WE_BYTES  = (size_t)kN * kD * 2;   // 32 MB  bf16 W_enc
  const size_t WDT_BYTES = (size_t)kN * kD * 4;   // 64 MB  W_dec^T
  const size_t IDX_BYTES = (size_t)kB * (2 * kK) * 4;
  const size_t VAL_BYTES = (size_t)kB * (2 * kK) * 4;
  __bf16* xb   = (__bf16*)ws;
  __bf16* web  = (__bf16*)(ws + XB_BYTES);
  float*  wdt  = (float*) (ws + XB_BYTES + WE_BYTES);
  int*    idxw = (int*)   (ws + XB_BYTES + WE_BYTES + WDT_BYTES);
  float*  valw = (float*) (ws + XB_BYTES + WE_BYTES + WDT_BYTES + IDX_BYTES);
  int*    cntw = (int*)   (ws + XB_BYTES + WE_BYTES + WDT_BYTES + IDX_BYTES + VAL_BYTES);

  sae_prep_xb<<<(kB * kD / 4) / 256, 256, 0, stream>>>(
      (const float4*)x, bv, (v4bf*)xb);
  sae_prep_wenc<<<((size_t)kN * kD / 4) / 256, 256, 0, stream>>>(
      (const float4*)We, (v4bf*)web);
  sae_transpose_wdec<<<dim3(kN / 32, kD / 32), dim3(32, 8), 0, stream>>>(Wd, wdt);
  sae_enc_gemm<<<dim3(kN / 128, kB / 128), 256, 0, stream>>>(xb, web, z);
  sae_topk<<<kB, 256, 0, stream>>>(z, idxw, valw, cntw);
  sae_recon<<<kB, 256, 0, stream>>>(wdt, bv, idxw, valw, cntw, recon);
}
